// MCPropNet_85023172591735
// MI455X (gfx1250) — matrix operand order repaired and checked by reading, not compiled, not run on hardware
//
#include <hip/hip_runtime.h>
#include <hip/hip_bf16.h>

typedef _Float16 h16;
typedef __attribute__((ext_vector_type(16))) _Float16     v16h;
typedef __attribute__((ext_vector_type(8)))  float        v8f;
typedef __attribute__((ext_vector_type(4)))  unsigned int v4u;

union Frag { v4u q[2]; v16h v; };   // 32 bytes each view

#define HH   512
#define WW   512
#define NB   4
#define HWSZ (HH*WW)
#define NPX  (NB*HWSZ)

__device__ __forceinline__ float sigmoidf_(float x){ return 1.0f/(1.0f+__expf(-x)); }

// ---------------------------------------------------------------------------
// K0: build Pn / edge / ML_f, pack conv inputs NHWC fp16 (padded to C=32),
//     initialize D0 and Dt.
// ---------------------------------------------------------------------------
__global__ __launch_bounds__(256) void prep_kernel(
    const float* __restrict__ I, const int* __restrict__ ML,
    const float* __restrict__ E, h16* __restrict__ xin, h16* __restrict__ pin,
    float* __restrict__ D0, float* __restrict__ dt0)
{
  int idx = blockIdx.x*256 + threadIdx.x;
  if (idx >= NPX) return;
  int b = idx / HWSZ; int s = idx - b*HWSZ; int r = s / WW; int c = s - r*WW;
  float e   = E[idx];
  float P   = fminf(fmaxf(e*10.f, 0.f), 10.f);
  float pn  = fminf(fmaxf(e, 0.f), 1.f);           // == clip(P/DMAX,0,1)
  float mlf = (ML[idx] > 0) ? 1.f : 0.f;
  float gx = 0.f, gy = 0.f;
  if (c > 0){ float el = fminf(fmaxf(E[idx-1 ], 0.f), 1.f); gx = pn - el; }
  if (r > 0){ float eu = fminf(fmaxf(E[idx-WW], 0.f), 1.f); gy = pn - eu; }
  float edge = fminf(fmaxf((fabsf(gx)+fabsf(gy))*0.5f, 0.f), 1.f);

  h16* xp = xin + (size_t)idx*32;
  h16* pp = pin + (size_t)idx*32;
  #pragma unroll
  for (int ch=0; ch<32; ++ch){ xp[ch]=(h16)0.f; pp[ch]=(h16)0.f; }
  xp[0]=(h16)I[(b*3+0)*HWSZ+s]; xp[1]=(h16)I[(b*3+1)*HWSZ+s]; xp[2]=(h16)I[(b*3+2)*HWSZ+s];
  xp[3]=(h16)pn; xp[4]=(h16)e; xp[5]=(h16)mlf;
  pp[0]=(h16)pn; pp[1]=(h16)edge; pp[2]=(h16)mlf;
  D0[idx]=P; dt0[idx]=P;
}

// ---------------------------------------------------------------------------
// K1: repack OIHW f32 3x3 weights into per-lane WMMA B-fragment fp16 layout.
// B element (k',N): lane = N%16 + 16*((k'/8)%2), half = k'%8 + 8*(k'/16) (K=32 chunk).
// Flat index: ((((tap*KC + kc)*4 + nt)*32) + lane)*16 + h
// ---------------------------------------------------------------------------
__global__ __launch_bounds__(256) void pack_w_kernel(
    const float* __restrict__ w, h16* __restrict__ out, int cin, int KC)
{
  int idx = blockIdx.x*256 + threadIdx.x;
  int total = KC*9*4*32*16;
  if (idx >= total) return;
  int hh   = idx & 15;
  int lane = (idx >> 4) & 31;
  int nt   = (idx >> 9) & 3;
  int rest = idx >> 11;
  int kc   = rest % KC;
  int tap  = rest / KC;
  int N  = nt*16 + (lane & 15);
  int kk = (hh & 7) + 8*(lane >> 4) + 16*(hh >> 3);
  int ci = kc*32 + kk;
  float v = 0.f;
  if (ci < cin) v = w[(N*cin + ci)*9 + tap];     // OIHW, tap = ky*3+kx
  out[idx] = (h16)v;
}

// ---------------------------------------------------------------------------
// K2: implicit-GEMM 3x3 conv + bias + ReLU via v_wmma_f32_16x16x32_f16.
// Block = 256 threads (8 waves). Tile = 8 rows x 16 cols of output pixels.
// Wave w computes row w: D(16px x 16ch) x 4 N-tiles, K = 9 taps * KC*32 ch.
// A-fragment layout: K = (h%8) + 16*(h/8) + 8*(lane>=16), M = lane%16.
// Both the input halo tile AND the packed weight fragments are staged in LDS
// (dynamic shared, ~97 KB for KC=2; 320 KB/WGP on CDNA5) so the steady-state
// inner loop is ds_load_b128 + v_wmma only (dscnt waits, no loadcnt drains).
// ---------------------------------------------------------------------------
template<int KC>
__global__ __launch_bounds__(256) void conv3x3_wmma_kernel(
    const h16* __restrict__ in, const h16* __restrict__ wpack,
    const float* __restrict__ bias, h16* __restrict__ out)
{
  constexpr int CIN = KC*32;
  constexpr int TILE_HALVES = 180*CIN;            // (8+2)x(16+2)xCIN fp16
  extern __shared__ __align__(16) char smem_raw[];
  h16* s_in = (h16*)smem_raw;
  v4u* s_w4 = (v4u*)(smem_raw + (size_t)TILE_HALVES*2);

  const int tid = threadIdx.x;
  const int x0 = blockIdx.x*16, y0 = blockIdx.y*8, b = blockIdx.z;

  // cooperative staging of packed weight fragments (KC*2304 x 16B)
  {
    const v4u* wp4 = (const v4u*)wpack;
    constexpr int W4 = KC*9*4*32*2;               // uint4 count
    for (int u = tid; u < W4; u += 256) s_w4[u] = wp4[u];
  }

  // cooperative staging of (8+2) x (16+2) x CIN fp16 input tile (zero-padded)
  const int chunks = 180*(CIN/8);
  for (int u = tid; u < chunks; u += 256){
    int pix = u / (CIN/8); int ch = u - pix*(CIN/8);
    int rr = pix/18, cc = pix - rr*18;
    int y = y0 + rr - 1, x = x0 + cc - 1;
    v4u val = {0u,0u,0u,0u};
    if (y>=0 && y<HH && x>=0 && x<WW)
      val = *(const v4u*)(in + ((size_t)((b*HH+y)*WW + x))*CIN + ch*8);
    *(v4u*)(&s_in[pix*CIN + ch*8]) = val;
  }
  __syncthreads();

  const int wave = tid >> 5, lane = tid & 31;
  const int mcol = lane & 15, g = lane >> 4;
  v8f acc[4] = {};

  #pragma unroll
  for (int tap=0; tap<9; ++tap){
    const int dy = tap/3, dx = tap - dy*3;
    #pragma unroll
    for (int kc=0; kc<KC; ++kc){
      Frag a;
      const h16* ap = &s_in[((wave+dy)*18 + (mcol+dx))*CIN + kc*32 + 8*g];
      a.q[0] = *(const v4u*)ap;          // K = 8g + 0..7
      a.q[1] = *(const v4u*)(ap + 16);   // K = 16 + 8g + 0..7
      #pragma unroll
      for (int nt=0; nt<4; ++nt){
        Frag bm;
        const v4u* bp = s_w4 + (((tap*KC + kc)*4 + nt)*32 + lane)*2;
        bm.q[0] = bp[0]; bm.q[1] = bp[1];
        acc[nt] = __builtin_amdgcn_wmma_f32_16x16x32_f16(
            false, a.v, false, bm.v, (short)0, acc[nt], false, false);
      }
    }
  }

  // epilogue: D[M][N], M = r + 8g (pixel col offset), N = mcol (channel)
  const int row = y0 + wave;
  #pragma unroll
  for (int nt=0; nt<4; ++nt){
    const int ch = nt*16 + mcol;
    const float bv = bias[ch];
    #pragma unroll
    for (int r=0; r<8; ++r){
      const int col = x0 + r + 8*g;
      float v = fmaxf(acc[nt][r] + bv, 0.f);
      out[((size_t)((b*HH+row)*WW + col))*64 + ch] = (h16)v;
    }
  }
}

// ---------------------------------------------------------------------------
// K3: curv/aff3/aff5 heads -> L1-normalized per-pixel coeffs (36 fp16/px).
// ---------------------------------------------------------------------------
__global__ __launch_bounds__(128) void coeff_kernel(
    const h16* __restrict__ pfeat, const float* __restrict__ E,
    const float* __restrict__ cw, const float* __restrict__ cb,
    const float* __restrict__ a3w, const float* __restrict__ a3b,
    const float* __restrict__ a5w, const float* __restrict__ a5b,
    h16* __restrict__ coeff)
{
  int idx = blockIdx.x*128 + threadIdx.x;
  if (idx >= NPX) return;
  union { v4u q[8]; h16 h[64]; } pf;
  const v4u* fp = (const v4u*)(pfeat + (size_t)idx*64);
  #pragma unroll
  for (int i=0;i<8;++i) pf.q[i] = fp[i];
  float pn = fminf(fmaxf(E[idx],0.f),1.f);
  float c0 = cb[0] + pn*cw[64];
  float c1 = cb[1] + pn*cw[129];
  #pragma unroll
  for (int c=0;c<64;++c){ float f=(float)pf.h[c]; c0 += f*cw[c]; c1 += f*cw[65+c]; }
  float scale = 0.1f + 0.9f*sigmoidf_(c0);   // KMIN + (KMAX-KMIN)*sigmoid
  float bv = c1;
  h16* op = coeff + (size_t)idx*36;
  {
    float l[9]; float sa=0.f;
    #pragma unroll
    for (int j=0;j<9;++j){
      float s = a3b[j];
      #pragma unroll
      for (int c=0;c<64;++c) s += (float)pf.h[c]*a3w[j*64+c];
      s = s*scale + bv; l[j]=s; sa += fabsf(s);
    }
    float inv = 1.f/(sa+1e-6f);
    #pragma unroll
    for (int j=0;j<9;++j) op[j] = (h16)(l[j]*inv);
  }
  {
    float l[25]; float sa=0.f;
    #pragma unroll
    for (int j=0;j<25;++j){
      float s = a5b[j];
      #pragma unroll
      for (int c=0;c<64;++c) s += (float)pf.h[c]*a5w[j*64+c];
      s = s*scale + bv; l[j]=s; sa += fabsf(s);
    }
    float inv = 1.f/(sa+1e-6f);
    #pragma unroll
    for (int j=0;j<25;++j) op[9+j] = (h16)(l[j]*inv);
  }
  op[34]=(h16)0.f; op[35]=(h16)0.f;
}

// ---------------------------------------------------------------------------
// K4: gate head -> 2-way softmax per pixel.
// ---------------------------------------------------------------------------
__global__ __launch_bounds__(256) void gate_kernel(
    const h16* __restrict__ feat, const float* __restrict__ gw,
    const float* __restrict__ gb, h16* __restrict__ sig)
{
  int idx = blockIdx.x*256 + threadIdx.x;
  if (idx >= NPX) return;
  union { v4u q[8]; h16 h[64]; } f;
  const v4u* fp = (const v4u*)(feat + (size_t)idx*64);
  #pragma unroll
  for (int i=0;i<8;++i) f.q[i] = fp[i];
  float g0 = gb[0], g1 = gb[1];
  #pragma unroll
  for (int c=0;c<64;++c){ float v=(float)f.h[c]; g0 += v*gw[c]; g1 += v*gw[64+c]; }
  float s0 = sigmoidf_(g0-g1);
  sig[(size_t)idx*2  ] = (h16)s0;
  sig[(size_t)idx*2+1] = (h16)(1.f-s0);
}

// ---------------------------------------------------------------------------
// K5: one propagation step. 16x16 tile + halo(2) in LDS; center -> D0.
// ---------------------------------------------------------------------------
__global__ __launch_bounds__(256) void prop_kernel(
    const float* __restrict__ dt_in, float* __restrict__ dt_out,
    const float* __restrict__ D0, const float* __restrict__ DL,
    const int* __restrict__ ML, const h16* __restrict__ coeff,
    const h16* __restrict__ sig)
{
  __shared__ float s_d[20*20];
  const int tid = threadIdx.x;
  const int x0 = blockIdx.x*16, y0 = blockIdx.y*16, b = blockIdx.z;
  for (int u=tid; u<400; u+=256){
    int rr=u/20, cc=u-rr*20;
    int y=y0+rr-2, x=x0+cc-2;
    float v = 0.f;
    if (y>=0 && y<HH && x>=0 && x<WW) v = dt_in[(size_t)b*HWSZ + y*WW + x];
    s_d[u] = v;
  }
  __syncthreads();
  const int tx = tid & 15, ty = tid >> 4;
  const size_t idx = (size_t)b*HWSZ + (size_t)(y0+ty)*WW + (x0+tx);
  const float d0 = D0[idx], dl = DL[idx];
  const float mlf = (ML[idx] > 0) ? 1.f : 0.f;
  const float s0 = (float)sig[idx*2], s1 = (float)sig[idx*2+1];
  const h16* cf = coeff + idx*36;
  const int ly = ty+2, lx = tx+2;
  float sum3 = 0.f;
  #pragma unroll
  for (int ky=0;ky<3;++ky)
    #pragma unroll
    for (int kx=0;kx<3;++kx){
      float v = (ky==1 && kx==1) ? d0 : s_d[(ly+ky-1)*20 + (lx+kx-1)];
      sum3 += (float)cf[ky*3+kx]*v;
    }
  float sum5 = 0.f;
  #pragma unroll
  for (int ky=0;ky<5;++ky)
    #pragma unroll
    for (int kx=0;kx<5;++kx){
      float v = (ky==2 && kx==2) ? d0 : s_d[(ly+ky-2)*20 + (lx+kx-2)];
      sum5 += (float)cf[9+ky*5+kx]*v;
    }
  float mix = s0*sum3 + s1*sum5;
  float o = (1.f-0.9f*mlf)*mix + 0.9f*mlf*dl;
  dt_out[idx] = fminf(fmaxf(o, 0.f), 10.f);
}

// ---------------------------------------------------------------------------
extern "C" void kernel_launch(void* const* d_in, const int* in_sizes, int n_in,
                              void* d_out, int out_size, void* d_ws, size_t ws_size,
                              hipStream_t stream)
{
  (void)in_sizes; (void)n_in; (void)out_size; (void)ws_size;
  const float* I      = (const float*)d_in[0];
  const float* DL     = (const float*)d_in[1];
  const int*   ML     = (const int*)  d_in[2];
  const float* E      = (const float*)d_in[3];
  const float* enc_w1 = (const float*)d_in[4];
  const float* enc_b1 = (const float*)d_in[5];
  const float* enc_w2 = (const float*)d_in[6];
  const float* enc_b2 = (const float*)d_in[7];
  const float* enc_w3 = (const float*)d_in[8];
  const float* enc_b3 = (const float*)d_in[9];
  const float* penc_w1= (const float*)d_in[10];
  const float* penc_b1= (const float*)d_in[11];
  const float* penc_w2= (const float*)d_in[12];
  const float* penc_b2= (const float*)d_in[13];
  const float* penc_w3= (const float*)d_in[14];
  const float* penc_b3= (const float*)d_in[15];
  const float* curv_w = (const float*)d_in[16];
  const float* curv_b = (const float*)d_in[17];
  const float* aff3_w = (const float*)d_in[18];
  const float* aff3_b = (const float*)d_in[19];
  const float* aff5_w = (const float*)d_in[20];
  const float* aff5_b = (const float*)d_in[21];
  const float* gate_w = (const float*)d_in[22];
  const float* gate_b = (const float*)d_in[23];

  char* ws = (char*)d_ws;
  size_t off = 0;
  auto alloc = [&](size_t bytes)->char*{
    char* p = ws + off; off += (bytes + 255) & ~(size_t)255; return p;
  };
  h16*   xin   = (h16*)  alloc((size_t)NPX*32*2);
  h16*   pin   = (h16*)  alloc((size_t)NPX*32*2);
  h16*   c0    = (h16*)  alloc((size_t)NPX*64*2);
  h16*   c1    = (h16*)  alloc((size_t)NPX*64*2);
  h16*   pfeat = (h16*)  alloc((size_t)NPX*64*2);   // reused later as feat
  h16*   coeff = (h16*)  alloc((size_t)NPX*36*2);
  h16*   sig   = (h16*)  alloc((size_t)NPX*2*2);
  float* D0    = (float*)alloc((size_t)NPX*4);
  float* dt0   = (float*)alloc((size_t)NPX*4);
  float* dt1   = (float*)alloc((size_t)NPX*4);
  h16*   wpE1  = (h16*)  alloc((size_t)1*9*4*32*16*2);
  h16*   wpE2  = (h16*)  alloc((size_t)2*9*4*32*16*2);
  h16*   wpE3  = (h16*)  alloc((size_t)2*9*4*32*16*2);
  h16*   wpP1  = (h16*)  alloc((size_t)1*9*4*32*16*2);
  h16*   wpP2  = (h16*)  alloc((size_t)2*9*4*32*16*2);
  h16*   wpP3  = (h16*)  alloc((size_t)2*9*4*32*16*2);

  dim3 blk(256);
  prep_kernel<<<dim3(NPX/256), blk, 0, stream>>>(I, ML, E, xin, pin, D0, dt0);
  pack_w_kernel<<<dim3(72),  blk, 0, stream>>>(enc_w1,  wpE1,  6, 1);
  pack_w_kernel<<<dim3(144), blk, 0, stream>>>(enc_w2,  wpE2, 64, 2);
  pack_w_kernel<<<dim3(144), blk, 0, stream>>>(enc_w3,  wpE3, 64, 2);
  pack_w_kernel<<<dim3(72),  blk, 0, stream>>>(penc_w1, wpP1,  3, 1);
  pack_w_kernel<<<dim3(144), blk, 0, stream>>>(penc_w2, wpP2, 64, 2);
  pack_w_kernel<<<dim3(144), blk, 0, stream>>>(penc_w3, wpP3, 64, 2);

  dim3 cgrid(WW/16, HH/8, NB);
  const size_t shb1 = (size_t)180*32*2 + (size_t)1*36864;   // KC=1: ~48 KB
  const size_t shb2 = (size_t)180*64*2 + (size_t)2*36864;   // KC=2: ~97 KB
  // penc chain -> p_feat, then coefficient heads
  conv3x3_wmma_kernel<1><<<cgrid, blk, shb1, stream>>>(pin, wpP1, penc_b1, c0);
  conv3x3_wmma_kernel<2><<<cgrid, blk, shb2, stream>>>(c0,  wpP2, penc_b2, c1);
  conv3x3_wmma_kernel<2><<<cgrid, blk, shb2, stream>>>(c1,  wpP3, penc_b3, pfeat);
  coeff_kernel<<<dim3(NPX/128), dim3(128), 0, stream>>>(
      pfeat, E, curv_w, curv_b, aff3_w, aff3_b, aff5_w, aff5_b, coeff);
  // enc chain -> feat (reuse pfeat buffer), then gate
  conv3x3_wmma_kernel<1><<<cgrid, blk, shb1, stream>>>(xin, wpE1, enc_b1, c0);
  conv3x3_wmma_kernel<2><<<cgrid, blk, shb2, stream>>>(c0,  wpE2, enc_b2, c1);
  conv3x3_wmma_kernel<2><<<cgrid, blk, shb2, stream>>>(c1,  wpE3, enc_b3, pfeat);
  gate_kernel<<<dim3(NPX/256), blk, 0, stream>>>(pfeat, gate_w, gate_b, sig);

  // 6 propagation steps, ping-pong, last writes d_out (f32)
  dim3 pgrid(WW/16, HH/16, NB);
  for (int s=0; s<6; ++s){
    const float* src = (s & 1) ? dt1 : dt0;
    float* dst = (s == 5) ? (float*)d_out : ((s & 1) ? dt0 : dt1);
    prop_kernel<<<pgrid, blk, 0, stream>>>(src, dst, D0, DL, ML, coeff, sig);
  }
}
